// SuperPointDINO_72395968741565
// MI455X (gfx1250) — matrix operand order repaired and checked by reading, not compiled, not run on hardware
//
#include <hip/hip_runtime.h>
#include <cstdint>
#include <cstddef>

// ---------------- problem constants ----------------
#define BZ    8
#define CIN   256
#define HC    120
#define WC    160
#define HWc   (HC*WC)          // 19200
#define C5c   256
#define DETH  65
#define KTOP  1024
#define HF    (HC*8)           // 960
#define WF    (WC*8)           // 1280
#define HWf   (HF*WF)          // 1228800
#define CAND  4096
#define NBIN  4096

typedef __attribute__((ext_vector_type(16))) __bf16 v16bf;
typedef __attribute__((ext_vector_type(8)))  float  v8f;

static __device__ __forceinline__ unsigned short bf16bits(float f) {
    return __builtin_bit_cast(unsigned short, (__bf16)f);
}

// =====================================================================
// prep: fp32 -> bf16 bulk convert (4 elems/thread, b128 in / b64 out)
// =====================================================================
__global__ void cvt_f32_bf16_vec4(const float* __restrict__ in,
                                  unsigned short* __restrict__ out, int n4)
{
    int i = blockIdx.x*256 + threadIdx.x;
    if (i >= n4) return;
    float4 f = ((const float4*)in)[i];
    ushort4 o;
    o.x = bf16bits(f.x); o.y = bf16bits(f.y);
    o.z = bf16bits(f.z); o.w = bf16bits(f.w);
    ((ushort4*)out)[i] = o;
}

// prep: (co,ci,kh,kw) f32 -> [tap][co][ci] bf16
__global__ void prep_w3(const float* __restrict__ w, unsigned short* __restrict__ out)
{
    int i = blockIdx.x*256 + threadIdx.x;
    if (i >= 9*CIN*CIN) return;
    int tap = i / (CIN*CIN);
    int r   = i % (CIN*CIN);
    int co  = r / CIN, ci = r % CIN;
    out[i] = bf16bits(w[((size_t)co*CIN + ci)*9 + tap]);
}

// prep: (oc,ci) f32 -> [256 padded oc][ci] bf16, zero-filled past OC
__global__ void prep_w1(const float* __restrict__ w, unsigned short* __restrict__ out, int OC)
{
    int i = blockIdx.x*256 + threadIdx.x;
    if (i >= 256*CIN) return;
    int co = i / CIN, ci = i % CIN;
    float v = (co < OC) ? w[(size_t)co*CIN + ci] : 0.f;
    out[i] = bf16bits(v);
}

// =====================================================================
// 3x3 conv (pad 1) + bias + BN + ReLU, bf16 WMMA implicit GEMM.
// 256 thr = 8 waves (4 co-subtiles x 2 pix-subtiles). Tile 64co x 32pix.
// Per (kh, cc): async-load 3 taps of weights to LDS (plain [tap][co][ci] order),
// stage a 34-wide activation halo, then 3 WMMAs (one per kw).
// Output written as bf16 (only consumed by the 1x1 WMMA).
// LDS: A = 3*64*32 bf16 @ 0, B = 2*34*16 bf16 @ 6144 (ushort units).
// =====================================================================
__global__ __launch_bounds__(256)
void conv3x3_bn_wmma(const unsigned short* __restrict__ xbf,
                     const unsigned short* __restrict__ wT3,  // [9][256][256]
                     const float* __restrict__ bias, const float* __restrict__ bsc,
                     const float* __restrict__ bsh, unsigned short* __restrict__ ybf)
{
    extern __shared__ unsigned short sm[];
    const int B_OFF = 3*64*32;   // 6144

    const int tid    = threadIdx.x;
    const int lane   = tid & 31;
    const int wave   = tid >> 5;
    const int coSub  = wave >> 1;
    const int pixSub = wave & 1;
    const int half   = lane >> 4;
    const int mrow   = lane & 15;

    const int b       = blockIdx.z;
    const int coBase  = blockIdx.y * 64;
    const int pixBase = blockIdx.x * 32;     // 160 % 32 == 0: never crosses rows
    const int h       = pixBase / WC;
    const int w0      = pixBase % WC;

    // activation staging assignment: 32 ci x 8 groups of 4 pixels
    const int skk = tid >> 3;          // ci within chunk 0..31
    const int sp4 = (tid & 7) * 4;     // pixel group base 0..28

    v8f acc = {0.f,0.f,0.f,0.f,0.f,0.f,0.f,0.f};

    #pragma unroll 1
    for (int kh = 0; kh < 3; ++kh) {
        const int  hh    = h + kh - 1;
        const bool rowOK = (hh >= 0) && (hh < HC);
        const unsigned short* wbase = wT3 + (size_t)(kh*3)*CIN*CIN;  // uniform

        #pragma unroll 1
        for (int cc = 0; cc < CIN; cc += 32) {
            // ---- async weight staging: 3 taps x 64co x 32ci = 768 x 16B ----
            #pragma unroll
            for (int g = 0; g < 3; ++g) {
                int e    = g*256 + tid;
                int tap  = e >> 8;
                int rem  = e & 255;
                int co   = rem >> 2;
                int grp  = rem & 3;
                int pgrp = grp;   // {0,1,2,3}->{0,1,2,3}
                unsigned goff = (unsigned)((((tap*CIN) + coBase + co)*CIN + cc + grp*8) * 2);
                unsigned loff = (unsigned)((((tap*64) + co)*32 + pgrp*8) * 2);
                asm volatile("global_load_async_to_lds_b128 %0, %1, %2 offset:0"
                             :: "v"(loff), "v"(goff), "s"(wbase) : "memory");
            }

            // ---- activation halo staging: [khalf][34 pix][16] ----
            {
                ushort4 vv = {0,0,0,0};
                if (rowOK)
                    vv = *(const ushort4*)(xbf + ((size_t)b*CIN + cc + skk)*HWc
                                               + (size_t)hh*WC + w0 + sp4);
                int hk = skk >> 4, kl = skk & 15;
                sm[B_OFF + (hk*34 + 1 + sp4 + 0)*16 + kl] = vv.x;
                sm[B_OFF + (hk*34 + 1 + sp4 + 1)*16 + kl] = vv.y;
                sm[B_OFF + (hk*34 + 1 + sp4 + 2)*16 + kl] = vv.z;
                sm[B_OFF + (hk*34 + 1 + sp4 + 3)*16 + kl] = vv.w;
            }
            if (tid < 64) {   // halo columns (pix 0 and 33)
                int kk   = tid >> 1;
                int side = tid & 1;
                int pix  = side ? 33 : 0;
                int col  = w0 - 1 + pix;
                unsigned short v = 0;
                if (rowOK && col >= 0 && col < WC)
                    v = xbf[((size_t)b*CIN + cc + kk)*HWc + (size_t)hh*WC + col];
                sm[B_OFF + ((kk >> 4)*34 + pix)*16 + (kk & 15)] = v;
            }

            asm volatile("s_wait_asynccnt 0" ::: "memory");
            __syncthreads();

            // ---- 3 WMMAs: one per kw, shifted halo window ----
            #pragma unroll
            for (int kw = 0; kw < 3; ++kw) {
                v16bf af = *(const v16bf*)(sm + ((kw*64 + coSub*16 + mrow)*32 + half*16));
                int ci2 = kw + pixSub*16 + mrow;
                v16bf bfv = *(const v16bf*)(sm + B_OFF + (half*34 + ci2)*16);
                acc = __builtin_amdgcn_wmma_f32_16x16x32_bf16(
                          false, af, false, bfv, (short)0, acc, false, false);
            }
            __syncthreads();
        }
    }

    const int p = pixBase + pixSub*16 + mrow;
    #pragma unroll
    for (int r = 0; r < 8; ++r) {
        int m  = r + (half << 3);
        int co = coBase + coSub*16 + m;
        float val = (acc[r] + bias[co]) * bsc[co] + bsh[co];
        val = fmaxf(val, 0.f);
        ybf[((size_t)b*C5c + co)*HWc + p] = bf16bits(val);
    }
}

// =====================================================================
// 1x1 conv + bias + BN, bf16 WMMA GEMM. K=256.
// Weights from padded [256][256] bf16 (async to LDS); acts bf16.
// LDS: A = 64*32 @ 0, B = [2][32][16] @ 2048 (ushort units).
// =====================================================================
__global__ __launch_bounds__(256)
void conv1x1_bn_wmma(const unsigned short* __restrict__ xbf,
                     const unsigned short* __restrict__ w1p,  // [256][256] bf16
                     const float* __restrict__ bias, const float* __restrict__ bsc,
                     const float* __restrict__ bsh, float* __restrict__ y, int OC)
{
    extern __shared__ unsigned short sm[];
    const int B_OFF = 64*32;   // 2048

    const int tid    = threadIdx.x;
    const int lane   = tid & 31;
    const int wave   = tid >> 5;
    const int coSub  = wave >> 1;
    const int pixSub = wave & 1;
    const int half   = lane >> 4;
    const int mrow   = lane & 15;

    const int b       = blockIdx.z;
    const int coBase  = blockIdx.y * 64;
    const int pixBase = blockIdx.x * 32;

    const int skk = tid >> 3;
    const int sp4 = (tid & 7) * 4;

    v8f acc = {0.f,0.f,0.f,0.f,0.f,0.f,0.f,0.f};

    #pragma unroll 1
    for (int cc = 0; cc < CIN; cc += 32) {
        // async weight staging: 64co x 32ci = 256 x 16B granules, 1/thread
        {
            int co   = tid >> 2;
            int grp  = tid & 3;
            int pgrp = grp;
            unsigned goff = (unsigned)((((coBase + co)*CIN) + cc + grp*8) * 2);
            unsigned loff = (unsigned)((co*32 + pgrp*8) * 2);
            asm volatile("global_load_async_to_lds_b128 %0, %1, %2 offset:0"
                         :: "v"(loff), "v"(goff), "s"(w1p) : "memory");
        }
        // activation staging: [khalf][32 pix][16]
        {
            size_t gaddr = ((size_t)b*CIN + cc + skk)*HWc + pixBase + sp4;
            ushort4 vv = *(const ushort4*)(xbf + gaddr);
            int hk = skk >> 4, kl = skk & 15;
            sm[B_OFF + (hk*32 + sp4 + 0)*16 + kl] = vv.x;
            sm[B_OFF + (hk*32 + sp4 + 1)*16 + kl] = vv.y;
            sm[B_OFF + (hk*32 + sp4 + 2)*16 + kl] = vv.z;
            sm[B_OFF + (hk*32 + sp4 + 3)*16 + kl] = vv.w;
            if (cc + 32 < CIN)
                __builtin_prefetch(xbf + gaddr + (size_t)32*HWc, 0, 0);
        }

        asm volatile("s_wait_asynccnt 0" ::: "memory");
        __syncthreads();

        v16bf af  = *(const v16bf*)(sm + ((coSub*16 + mrow)*32 + half*16));
        v16bf bfv = *(const v16bf*)(sm + B_OFF + (half*32 + pixSub*16 + mrow)*16);
        acc = __builtin_amdgcn_wmma_f32_16x16x32_bf16(
                  false, af, false, bfv, (short)0, acc, false, false);
        __syncthreads();
    }

    const int p = pixBase + pixSub*16 + mrow;
    #pragma unroll
    for (int r = 0; r < 8; ++r) {
        int m  = r + (half << 3);
        int co = coBase + coSub*16 + m;
        if (co < OC) {
            float val = (acc[r] + bias[co]) * bsc[co] + bsh[co];
            y[((size_t)b*OC + co)*HWc + p] = val;
        }
    }
}

// =====================================================================
// softmax over 65 channels + drop dustbin + 8x8 pixel shuffle
// =====================================================================
__global__ void softmax_shuffle(const float* __restrict__ semi, float* __restrict__ sc)
{
    int idx = blockIdx.x*256 + threadIdx.x;
    if (idx >= BZ*HWc) return;
    int b = idx / HWc, p = idx % HWc;
    int h = p / WC, w = p % WC;
    const float* s = semi + (size_t)b*DETH*HWc + p;
    float mx = -1e30f;
    for (int c = 0; c < DETH; ++c) mx = fmaxf(mx, s[(size_t)c*HWc]);
    float sum = 0.f;
    for (int c = 0; c < DETH; ++c) sum += __expf(s[(size_t)c*HWc] - mx);
    float inv = 1.f / sum;
    for (int r = 0; r < 8; ++r)
        for (int ccol = 0; ccol < 8; ++ccol) {
            float ev = __expf(s[(size_t)(r*8+ccol)*HWc] - mx) * inv;
            sc[(size_t)b*HWf + (size_t)(h*8+r)*WF + (w*8+ccol)] = ev;
        }
}

// ---------------- separable 9-tap max pool ----------------
__global__ void hmax9(const float* __restrict__ in, float* __restrict__ out)
{
    int idx = blockIdx.x*256 + threadIdx.x;
    if (idx >= BZ*HWf) return;
    int b = idx / HWf, r = idx % HWf;
    int yy = r / WF, xx = r % WF;
    int x0 = max(xx-4, 0), x1 = min(xx+4, WF-1);
    const float* row = in + (size_t)b*HWf + (size_t)yy*WF;
    float m = -1e30f;
    for (int xc = x0; xc <= x1; ++xc) m = fmaxf(m, row[xc]);
    out[idx] = m;
}

__global__ void vmax9(const float* __restrict__ in, float* __restrict__ out)
{
    int idx = blockIdx.x*256 + threadIdx.x;
    if (idx >= BZ*HWf) return;
    int b = idx / HWf, r = idx % HWf;
    int yy = r / WF, xx = r % WF;
    int y0 = max(yy-4, 0), y1 = min(yy+4, HF-1);
    const float* col = in + (size_t)b*HWf + xx;
    float m = -1e30f;
    for (int yc = y0; yc <= y1; ++yc) m = fmaxf(m, col[(size_t)yc*WF]);
    out[idx] = m;
}

__global__ void eq_mask(const float* a, const float* b, float* out)
{
    int idx = blockIdx.x*256 + threadIdx.x;
    if (idx >= BZ*HWf) return;
    out[idx] = (a[idx] == b[idx]) ? 1.f : 0.f;
}

__global__ void nms_suppress(const float* mpmask, const float* sc, float* out)
{
    int idx = blockIdx.x*256 + threadIdx.x;
    if (idx >= BZ*HWf) return;
    out[idx] = (mpmask[idx] > 0.f) ? 0.f : sc[idx];
}

__global__ void nms_combine(float* msk, const float* supp_sc, const float* mp_supp,
                            const float* mpmask)
{
    int idx = blockIdx.x*256 + threadIdx.x;
    if (idx >= BZ*HWf) return;
    float newmax = (supp_sc[idx] == mp_supp[idx]) ? 1.f : 0.f;
    float supp   = (mpmask[idx] > 0.f) ? 1.f : 0.f;
    msk[idx] = fmaxf(msk[idx], newmax * (1.f - supp));
}

__global__ void nms_final(const float* msk, const float* sc, float* sm)
{
    int idx = blockIdx.x*256 + threadIdx.x;
    if (idx >= BZ*HWf) return;
    int r = idx % HWf;
    int yy = r / WF, xx = r % WF;
    float v = (msk[idx] > 0.f) ? sc[idx] : 0.f;
    bool border = (xx >= 4) && (xx < WF-4) && (yy >= 4) && (yy < HF-4);
    sm[idx] = (v > 0.005f && border) ? v : 0.f;
}

// ---------------- top-k: histogram threshold + collect + bitonic ----------------
__global__ void zero_ints(int* p, int n)
{
    int idx = blockIdx.x*256 + threadIdx.x;
    if (idx < n) p[idx] = 0;
}

__global__ void topk_hist(const float* __restrict__ sm, int* __restrict__ hist)
{
    int idx = blockIdx.x*256 + threadIdx.x;
    if (idx >= BZ*HWf) return;
    int b = idx / HWf;
    unsigned bits = __float_as_uint(sm[idx]);
    atomicAdd(&hist[b*NBIN + (int)(bits >> 20)], 1);
}

__global__ void topk_thresh(const int* __restrict__ hist, int* __restrict__ binT)
{
    if (threadIdx.x != 0) return;
    int b = blockIdx.x;
    int cum = 0, sel = 0;
    for (int bin = NBIN-1; bin >= 0; --bin) {
        cum += hist[b*NBIN + bin];
        if (cum >= KTOP) { sel = bin; break; }
    }
    binT[b] = sel;
}

__global__ void topk_collect(const float* __restrict__ sm, const int* __restrict__ binT,
                             int* __restrict__ cnt, float* __restrict__ candV,
                             int* __restrict__ candI)
{
    int idx = blockIdx.x*256 + threadIdx.x;
    if (idx >= BZ*HWf) return;
    int b = idx / HWf, p = idx % HWf;
    float v = sm[idx];
    int bin = (int)(__float_as_uint(v) >> 20);
    if (bin >= binT[b]) {
        int pos = atomicAdd(&cnt[b], 1);
        if (pos < CAND) { candV[b*CAND + pos] = v; candI[b*CAND + pos] = p; }
    }
}

__global__ __launch_bounds__(1024)
void topk_sort(const int* __restrict__ cnt, const float* __restrict__ candV,
               const int* __restrict__ candI, float* __restrict__ outKp,
               float* __restrict__ outVals)
{
    __shared__ float sv[CAND];
    __shared__ int   si[CAND];
    const int b = blockIdx.x, t = threadIdx.x;
    int n = cnt[b]; if (n > CAND) n = CAND;

    for (int j = t; j < CAND; j += 1024) {
        if (j < n) { sv[j] = candV[b*CAND + j]; si[j] = candI[b*CAND + j]; }
        else       { sv[j] = -1.f;              si[j] = 0x7fffffff; }
    }
    __syncthreads();

    for (int k = 2; k <= CAND; k <<= 1) {
        for (int j = k >> 1; j > 0; j >>= 1) {
            for (int base = 0; base < CAND; base += 1024) {
                int i = base + t;
                int l = i ^ j;
                if (l > i) {
                    float vi = sv[i], vl = sv[l];
                    int   ii = si[i], il = si[l];
                    bool iRanksLower = (vi < vl) || (vi == vl && ii > il);
                    bool doSwap = ((i & k) == 0) ? iRanksLower : !iRanksLower;
                    if (doSwap) { sv[i] = vl; sv[l] = vi; si[i] = il; si[l] = ii; }
                }
            }
            __syncthreads();
        }
    }

    if (t < KTOP) {
        float v = sv[t]; int p = si[t];
        if (p == 0x7fffffff) { v = 0.f; p = 0; }
        outVals[b*KTOP + t] = v;
        outKp[(b*KTOP + t)*2 + 0] = (float)(p % WF);
        outKp[(b*KTOP + t)*2 + 1] = (float)(p / WF);
    }
}

// ---------------- descriptor L2 normalize (per pixel, 256 ch) ----------------
__global__ void desc_norm(float* __restrict__ d)
{
    int idx = blockIdx.x*256 + threadIdx.x;
    if (idx >= BZ*HWc) return;
    int b = idx / HWc, p = idx % HWc;
    float* base = d + (size_t)b*CIN*HWc + p;
    float s = 0.f;
    for (int c = 0; c < CIN; ++c) { float v = base[(size_t)c*HWc]; s += v*v; }
    float inv = 1.f / fmaxf(sqrtf(s), 1e-12f);
    for (int c = 0; c < CIN; ++c) base[(size_t)c*HWc] *= inv;
}

// ---------------- bilinear sample + renormalize ----------------
__global__ __launch_bounds__(256)
void sample_desc(const float* __restrict__ kp, const float* __restrict__ desc,
                 float* __restrict__ out)
{
    const int b = blockIdx.y, kq = blockIdx.x, c = threadIdx.x;
    float kx = kp[((size_t)b*KTOP + kq)*2 + 0];
    float ky = kp[((size_t)b*KTOP + kq)*2 + 1];
    float ux = (kx - 3.5f) / ((float)(WC*8) - 4.5f);
    float uy = (ky - 3.5f) / ((float)(HC*8) - 4.5f);
    float ix = ux * (float)(WC - 1);
    float iy = uy * (float)(HC - 1);
    float fx0 = floorf(ix), fy0 = floorf(iy);
    float wx = ix - fx0, wy = iy - fy0;
    int ix0 = min(max((int)fx0, 0), WC-1);
    int iy0 = min(max((int)fy0, 0), HC-1);
    int ix1 = min(ix0 + 1, WC-1);
    int iy1 = min(iy0 + 1, HC-1);

    const float* dc = desc + ((size_t)b*CIN + c)*HWc;
    float v = dc[iy0*WC+ix0]*(1.f-wx)*(1.f-wy) + dc[iy0*WC+ix1]*wx*(1.f-wy)
            + dc[iy1*WC+ix0]*(1.f-wx)*wy       + dc[iy1*WC+ix1]*wx*wy;

    __shared__ float red[256];
    red[c] = v*v;
    __syncthreads();
    for (int s = 128; s > 0; s >>= 1) {
        if (c < s) red[c] += red[c+s];
        __syncthreads();
    }
    float inv = 1.f / fmaxf(sqrtf(red[0]), 1e-12f);
    out[((size_t)b*CIN + c)*KTOP + kq] = v * inv;
}

// =====================================================================
extern "C" void kernel_launch(void* const* d_in, const int* in_sizes, int n_in,
                              void* d_out, int out_size, void* d_ws, size_t ws_size,
                              hipStream_t stream)
{
    (void)in_sizes; (void)n_in; (void)out_size; (void)ws_size;
    const float* x4  = (const float*)d_in[0];
    const float* wPa = (const float*)d_in[1];
    const float* bPa = (const float*)d_in[2];
    const float* sPa = (const float*)d_in[3];
    const float* tPa = (const float*)d_in[4];
    const float* wPb = (const float*)d_in[5];
    const float* bPb = (const float*)d_in[6];
    const float* sPb = (const float*)d_in[7];
    const float* tPb = (const float*)d_in[8];
    const float* wDa = (const float*)d_in[9];
    const float* bDa = (const float*)d_in[10];
    const float* sDa = (const float*)d_in[11];
    const float* tDa = (const float*)d_in[12];
    const float* wDb = (const float*)d_in[13];
    const float* bDb = (const float*)d_in[14];
    const float* sDb = (const float*)d_in[15];
    const float* tDb = (const float*)d_in[16];

    float* out     = (float*)d_out;
    float* outKp   = out;                          // (8,1024,2)
    float* outVals = out + (size_t)BZ*KTOP*2;      // (8,1024)
    float* outDesc = out + (size_t)BZ*KTOP*3;      // (8,256,1024)

    char* ws = (char*)d_ws;
    size_t off = 0;
    auto carve = [&](size_t bytes) -> char* {
        char* p = ws + off;
        off += (bytes + 255) & ~(size_t)255;
        return p;
    };
    unsigned short* xbf     = (unsigned short*)carve((size_t)BZ*CIN*HWc*2);
    unsigned short* featAbf = (unsigned short*)carve((size_t)BZ*CIN*HWc*2);
    unsigned short* wT3     = (unsigned short*)carve((size_t)9*CIN*CIN*2);
    unsigned short* w1p     = (unsigned short*)carve((size_t)256*CIN*2);
    float* featB = (float*)carve((size_t)BZ*CIN*HWc*4);
    float* semi  = (float*)carve((size_t)BZ*DETH*HWc*4);
    float* sc    = (float*)carve((size_t)BZ*HWf*4);
    float* msk   = (float*)carve((size_t)BZ*HWf*4);
    float* t1    = (float*)carve((size_t)BZ*HWf*4);
    float* t2    = (float*)carve((size_t)BZ*HWf*4);
    float* t3    = (float*)carve((size_t)BZ*HWf*4);
    float* t4    = (float*)carve((size_t)BZ*HWf*4);
    int*   hist  = (int*)carve((size_t)(BZ*NBIN + BZ)*4);
    int*   cnt   = hist + BZ*NBIN;
    int*   binT  = (int*)carve(BZ*4);
    float* candV = (float*)carve((size_t)BZ*CAND*4);
    int*   candI = (int*)carve((size_t)BZ*CAND*4);

    const dim3 blk(256);
    const dim3 g3(HWc/32, C5c/64, BZ);                 // 600,4,8
    const int  mapBlocks    = (BZ*HWf + 255)/256;
    const int  coarseBlocks = (BZ*HWc + 255)/256;
    const size_t shConv3 = (size_t)(3*64*32 + 2*34*16)*2;   // 14464 B
    const size_t shConv1 = (size_t)(64*32 + 2*32*16)*2;     // 6144 B

    // ---- prep: bf16 conversions / weight relayouts ----
    int n4 = BZ*CIN*HWc/4;
    cvt_f32_bf16_vec4<<<(n4+255)/256, blk, 0, stream>>>(x4, xbf, n4);
    prep_w3<<<(9*CIN*CIN+255)/256, blk, 0, stream>>>(wPa, wT3);

    // ---- detector branch ----
    conv3x3_bn_wmma<<<g3, blk, shConv3, stream>>>(xbf, wT3, bPa, sPa, tPa, featAbf);
    prep_w1<<<(256*CIN+255)/256, blk, 0, stream>>>(wPb, w1p, DETH);
    dim3 g1a(HWc/32, (DETH + 63)/64, BZ);
    conv1x1_bn_wmma<<<g1a, blk, shConv1, stream>>>(featAbf, w1p, bPb, sPb, tPb, semi, DETH);
    softmax_shuffle<<<coarseBlocks, blk, 0, stream>>>(semi, sc);

    // simple_nms
    hmax9<<<mapBlocks, blk, 0, stream>>>(sc, t1);
    vmax9<<<mapBlocks, blk, 0, stream>>>(t1, t2);
    eq_mask<<<mapBlocks, blk, 0, stream>>>(sc, t2, msk);
    for (int it = 0; it < 2; ++it) {
        hmax9<<<mapBlocks, blk, 0, stream>>>(msk, t1);
        vmax9<<<mapBlocks, blk, 0, stream>>>(t1, t2);
        nms_suppress<<<mapBlocks, blk, 0, stream>>>(t2, sc, t3);
        hmax9<<<mapBlocks, blk, 0, stream>>>(t3, t1);
        vmax9<<<mapBlocks, blk, 0, stream>>>(t1, t4);
        nms_combine<<<mapBlocks, blk, 0, stream>>>(msk, t3, t4, t2);
    }
    nms_final<<<mapBlocks, blk, 0, stream>>>(msk, sc, t1);       // t1 = sm

    // top-k
    zero_ints<<<(BZ*NBIN + BZ + 255)/256, blk, 0, stream>>>(hist, BZ*NBIN + BZ);
    topk_hist<<<mapBlocks, blk, 0, stream>>>(t1, hist);
    topk_thresh<<<BZ, 32, 0, stream>>>(hist, binT);
    topk_collect<<<mapBlocks, blk, 0, stream>>>(t1, binT, cnt, candV, candI);
    topk_sort<<<BZ, 1024, 0, stream>>>(cnt, candV, candI, outKp, outVals);

    // ---- descriptor branch ----
    prep_w3<<<(9*CIN*CIN+255)/256, blk, 0, stream>>>(wDa, wT3);
    conv3x3_bn_wmma<<<g3, blk, shConv3, stream>>>(xbf, wT3, bDa, sDa, tDa, featAbf);
    prep_w1<<<(256*CIN+255)/256, blk, 0, stream>>>(wDb, w1p, C5c);
    dim3 g1b(HWc/32, C5c/64, BZ);
    conv1x1_bn_wmma<<<g1b, blk, shConv1, stream>>>(featAbf, w1p, bDb, sDb, tDb, featB, C5c);
    desc_norm<<<coarseBlocks, blk, 0, stream>>>(featB);
    sample_desc<<<dim3(KTOP, BZ), blk, 0, stream>>>(outKp, featB, outDesc);
}
